// QueryInitial_35502199669121
// MI455X (gfx1250) — matrix operand order repaired and checked by reading, not compiled, not run on hardware
//
#include <hip/hip_runtime.h>
#include <hip/hip_bf16.h>
#include <math.h>

typedef __bf16 bf16_t;
typedef __attribute__((ext_vector_type(16))) __bf16 v16bf;
typedef __attribute__((ext_vector_type(8)))  __bf16 v8bf;
typedef __attribute__((ext_vector_type(8)))  float  v8f;

#define NPB   16384
#define BATCH 4
#define NTOT  (NPB * BATCH)   // 65536
#define CDIM  256
#define MAXNP 4096
#define NFPS  1024

// ---- flat output layout (floats), reference tuple order ----
#define OUT_VC    0                               // 4096*4      = 16384
#define OUT_VF    (OUT_VC + BATCH*NFPS*4)         // 4096*256    = 1048576
#define OUT_SEEDS (OUT_VF + BATCH*NFPS*CDIM)      // 65536*4     = 262144
#define OUT_SCLS  (OUT_SEEDS + NTOT*4)            // 65536*3     = 196608
#define OUT_VOTES (OUT_SCLS + NTOT*3)             // 65536*4     = 262144
#define OUT_FIDX  (OUT_VOTES + NTOT*4)            // 4096

// ------------------------------------------------------------------
// f32 -> bf16 conversion (weights only; X is converted in-register in GEMM)
// ------------------------------------------------------------------
__global__ void f32_to_bf16_kernel(const float* __restrict__ src,
                                   bf16_t* __restrict__ dst, int n) {
  int i = blockIdx.x * blockDim.x + threadIdx.x;
  if (i < n) dst[i] = (bf16_t)src[i];
}

__global__ void zero_kernel(float* __restrict__ p, int n) {
  int i = blockIdx.x * blockDim.x + threadIdx.x;
  if (i < n) p[i] = 0.0f;
}

// ------------------------------------------------------------------
// Fused dual-head GEMM:
//   H0 = X @ W0^T,  H1 = X @ W1^T   (X: NTOT x 256 f32, W: 256x256 (out,in) bf16)
// X is read ONCE as f32 and converted to bf16 A-fragments in registers;
// both L2-resident weight matrices reuse the same A fragments.
// Each wave: 16 rows x 256 cols, K=256 -> 8 v_wmma_f32_16x16x32_bf16 per tile,
// 2 heads x 16 tiles = 256 WMMA per wave. Fused per-column sum/sum^2 for BN.
// ------------------------------------------------------------------
__global__ __launch_bounds__(256)
void gemm_fused_kernel(const float* __restrict__ X,
                       const bf16_t* __restrict__ W0, const bf16_t* __restrict__ W1,
                       bf16_t* __restrict__ H0, bf16_t* __restrict__ H1,
                       float* __restrict__ stats /* [sum0|sq0|sum1|sq1] x 256 */) {
  __shared__ float ssum[2][CDIM];
  __shared__ float ssq[2][CDIM];
  int tid = threadIdx.x;
  ssum[0][tid] = 0.0f; ssum[1][tid] = 0.0f;
  ssq[0][tid]  = 0.0f; ssq[1][tid]  = 0.0f;
  __syncthreads();

  int wave = tid >> 5;
  int lane = tid & 31;
  int m    = lane & 15;
  int half = lane >> 4;
  long rowbase = ((long)blockIdx.x * 8 + wave) * 16;

  // A fragments (bf16 16x32 layout): lane L holds row (rowbase+m);
  // K = ks*32 + half*8 + {0..7} in v0..3 and +16 in v4..7.
  v16bf a[8];
  const float* xrow = X + (rowbase + m) * CDIM;
#pragma unroll
  for (int ks = 0; ks < 8; ++ks) {
    v8f f0 = *(const v8f*)(xrow + ks * 32 + half * 8);
    v8f f1 = *(const v8f*)(xrow + ks * 32 + 16 + half * 8);
    union { v16bf v; v8bf h[2]; } u;
    u.h[0] = __builtin_convertvector(f0, v8bf);
    u.h[1] = __builtin_convertvector(f1, v8bf);
    a[ks] = u.v;
  }

  int n = lane & 15;
#pragma unroll 1
  for (int hd = 0; hd < 2; ++hd) {
    const bf16_t* W = hd ? W1 : W0;
    bf16_t*       H = hd ? H1 : H0;
    for (int nt = 0; nt < 16; ++nt) {
      // B (32x16) layout: lane L holds col n; K = half*8 + {0..7} and +16.
      const bf16_t* wrow = W + (long)(nt * 16 + n) * CDIM;  // B[k][n] = W[n][k]
      v8f acc = {0.f, 0.f, 0.f, 0.f, 0.f, 0.f, 0.f, 0.f};
#pragma unroll
      for (int ks = 0; ks < 8; ++ks) {
        union { v16bf v; v8bf h[2]; } ub;
        ub.h[0] = *(const v8bf*)(wrow + ks * 32 + half * 8);
        ub.h[1] = *(const v8bf*)(wrow + ks * 32 + 16 + half * 8);
        acc = __builtin_amdgcn_wmma_f32_16x16x32_bf16(
            /*neg_a=*/false, a[ks], /*neg_b=*/false, ub.v,
            /*c_mod=*/(short)0, acc, /*reuse_a=*/false, /*reuse_b=*/false);
      }
      // D layout: VGPR r -> row (r + 8*half), lane&15 -> column
      float s = 0.0f, q = 0.0f;
#pragma unroll
      for (int r = 0; r < 8; ++r) {
        float v = acc[r];
        H[(rowbase + r + 8 * half) * CDIM + nt * 16 + n] = (bf16_t)v;
        s += v;
        q += v * v;
      }
      atomicAdd(&ssum[hd][nt * 16 + n], s);
      atomicAdd(&ssq[hd][nt * 16 + n], q);
    }
  }
  __syncthreads();
  atomicAdd(&stats[0   + tid], ssum[0][tid]);
  atomicAdd(&stats[256 + tid], ssq[0][tid]);
  atomicAdd(&stats[512 + tid], ssum[1][tid]);
  atomicAdd(&stats[768 + tid], ssq[1][tid]);
}

// ------------------------------------------------------------------
// BN stats finalize: stats[0..255]=sum_off [256..511]=sq_off
//                    [512..767]=sum_cls [768..1023]=sq_cls
// writes mu/rsig at [1024..2047]
// ------------------------------------------------------------------
__global__ void finalize_stats_kernel(float* __restrict__ stats) {
  int i = blockIdx.x * blockDim.x + threadIdx.x;
  if (i >= 512) return;
  int head = i >> 8, c = i & 255;
  float sum = stats[head * 512 + c];
  float sq  = stats[head * 512 + 256 + c];
  float mu  = sum / (float)NTOT;
  float var = sq / (float)NTOT - mu * mu;
  stats[1024 + head * 512 + c]       = mu;
  stats[1024 + head * 512 + 256 + c] = rsqrtf(var + 1e-5f);
}

// ------------------------------------------------------------------
// seeds = [bs, (x+0.5)*0.1*stride + 0, (y+0.5)*0.1*stride - 40, 0]
// ------------------------------------------------------------------
__global__ void seeds_kernel(const int* __restrict__ vc,
                             const int* __restrict__ stride_p,
                             float* __restrict__ seeds) {
  int n = blockIdx.x * blockDim.x + threadIdx.x;
  if (n >= NTOT) return;
  float st = (float)stride_p[0];
  int bs = vc[n * 3 + 0], y = vc[n * 3 + 1], x = vc[n * 3 + 2];
  seeds[n * 4 + 0] = (float)bs;
  seeds[n * 4 + 1] = ((float)x + 0.5f) * 0.1f * st + 0.0f;
  seeds[n * 4 + 2] = ((float)y + 0.5f) * 0.1f * st + (-40.0f);
  seeds[n * 4 + 3] = 0.0f;
}

// ------------------------------------------------------------------
// Head second halves: BN(train) -> ReLU -> Linear(C->3). One wave per row.
// ------------------------------------------------------------------
__device__ __forceinline__ void head2_row(const bf16_t* hrow,
                                          const float* mu, const float* rsig,
                                          const float* g, const float* bb,
                                          const float* W2, int lane,
                                          float& s0, float& s1, float& s2) {
  v8bf hv = *(const v8bf*)(hrow + lane * 8);
  s0 = 0.f; s1 = 0.f; s2 = 0.f;
#pragma unroll
  for (int j = 0; j < 8; ++j) {
    int c = lane * 8 + j;
    float h = (float)hv[j];
    h = (h - mu[c]) * rsig[c] * g[c] + bb[c];
    h = fmaxf(h, 0.0f);
    s0 += h * W2[c];
    s1 += h * W2[256 + c];
    s2 += h * W2[512 + c];
  }
#pragma unroll
  for (int off = 16; off > 0; off >>= 1) {
    s0 += __shfl_down(s0, off, 32);
    s1 += __shfl_down(s1, off, 32);
    s2 += __shfl_down(s2, off, 32);
  }
}

__global__ __launch_bounds__(256)
void head2_off_kernel(const bf16_t* __restrict__ H,
                      const float* __restrict__ mu, const float* __restrict__ rsig,
                      const float* __restrict__ g, const float* __restrict__ bb,
                      const float* __restrict__ W2, const float* __restrict__ b2,
                      const float* __restrict__ seeds, float* __restrict__ votes) {
  int wave = threadIdx.x >> 5, lane = threadIdx.x & 31;
  long row = (long)blockIdx.x * 8 + wave;
  float s0, s1, s2;
  head2_row(H + row * CDIM, mu, rsig, g, bb, W2, lane, s0, s1, s2);
  if (lane == 0) {
    float o0 = fminf(fmaxf(s0 + b2[0], -3.0f), 3.0f);
    float o1 = fminf(fmaxf(s1 + b2[1], -3.0f), 3.0f);
    float o2 = fminf(fmaxf(s2 + b2[2], -2.0f), 2.0f);
    votes[row * 4 + 0] = seeds[row * 4 + 0];
    votes[row * 4 + 1] = seeds[row * 4 + 1] + o0;
    votes[row * 4 + 2] = seeds[row * 4 + 2] + o1;
    votes[row * 4 + 3] = seeds[row * 4 + 3] + o2;
  }
}

__global__ __launch_bounds__(256)
void head2_cls_kernel(const bf16_t* __restrict__ H,
                      const float* __restrict__ mu, const float* __restrict__ rsig,
                      const float* __restrict__ g, const float* __restrict__ bb,
                      const float* __restrict__ W2, const float* __restrict__ b2,
                      float* __restrict__ seed_cls, float* __restrict__ sb) {
  int wave = threadIdx.x >> 5, lane = threadIdx.x & 31;
  long row = (long)blockIdx.x * 8 + wave;
  float s0, s1, s2;
  head2_row(H + row * CDIM, mu, rsig, g, bb, W2, lane, s0, s1, s2);
  if (lane == 0) {
    float c0 = s0 + b2[0], c1 = s1 + b2[1], c2 = s2 + b2[2];
    seed_cls[row * 3 + 0] = c0;
    seed_cls[row * 3 + 1] = c1;
    seed_cls[row * 3 + 2] = c2;
    float sc = fmaxf(c0, fmaxf(c1, c2));
    sb[row] = 1.0f / (1.0f + expf(-sc));
  }
}

// ------------------------------------------------------------------
// Per-batch descending bitonic sort of 16384 (score,idx) keys in LDS (128 KB,
// fits CDNA5 320KB/WGP LDS). Ties break toward lower index (matches lax.top_k).
// ------------------------------------------------------------------
__global__ __launch_bounds__(1024)
void topk_kernel(const float* __restrict__ sb, int* __restrict__ topk) {
  __shared__ unsigned long long key[NPB];   // 128 KB
  int b = blockIdx.x, tid = threadIdx.x;
  for (int i = tid; i < NPB; i += 1024) {
    unsigned int bits = __float_as_uint(sb[b * NPB + i]);
    unsigned int ord = (bits & 0x80000000u) ? ~bits : (bits | 0x80000000u);
    key[i] = ((unsigned long long)ord << 32) | (unsigned long long)(NPB - 1 - i);
  }
  __syncthreads();
  for (int k = 2; k <= NPB; k <<= 1) {
    for (int j = k >> 1; j > 0; j >>= 1) {
      for (int i = tid; i < NPB; i += 1024) {
        int ixj = i ^ j;
        if (ixj > i) {
          bool descRegion = ((i & k) == 0);
          unsigned long long a = key[i], c = key[ixj];
          if (descRegion ? (a < c) : (a > c)) { key[i] = c; key[ixj] = a; }
        }
      }
      __syncthreads();
    }
  }
  for (int t = tid; t < MAXNP; t += 1024)
    topk[b * MAXNP + t] = NPB - 1 - (int)(key[t] & 0xFFFFFFFFull);
}

// ------------------------------------------------------------------
// FPS: pool + distances resident in LDS; argmax key packs (dist, ~idx) so the
// max picks first-max index like jnp.argmax. idxs[0] = 0 per scan semantics.
// ------------------------------------------------------------------
__global__ __launch_bounds__(1024)
void fps_kernel(const int* __restrict__ topk, const float* __restrict__ seeds,
                int* __restrict__ pool) {
  __shared__ float px[MAXNP], py[MAXNP], pz[MAXNP], dist[MAXNP];
  __shared__ unsigned long long red[1024];
  __shared__ int sfar;
  int b = blockIdx.x, tid = threadIdx.x;
  for (int i = tid; i < MAXNP; i += 1024) {
    int li = topk[b * MAXNP + i];
    long r = (long)b * NPB + li;
    px[i] = seeds[r * 4 + 1];
    py[i] = seeds[r * 4 + 2];
    pz[i] = seeds[r * 4 + 3];
    dist[i] = 1e10f;
  }
  if (tid == 0) { sfar = 0; pool[b * NFPS + 0] = 0; }
  __syncthreads();
  for (int t = 1; t < NFPS; ++t) {
    int far = sfar;
    float fx = px[far], fy = py[far], fz = pz[far];
    unsigned long long best = 0ull;
    for (int i = tid; i < MAXNP; i += 1024) {
      float dx = px[i] - fx, dy = py[i] - fy, dz = pz[i] - fz;
      float d = dx * dx + dy * dy + dz * dz;
      float dm = fminf(dist[i], d);
      dist[i] = dm;
      unsigned long long kk =
          ((unsigned long long)__float_as_uint(dm) << 32) |
          (unsigned long long)(MAXNP - 1 - i);
      best = (kk > best) ? kk : best;
    }
    red[tid] = best;
    __syncthreads();
    for (int off = 512; off > 0; off >>= 1) {
      if (tid < off) {
        unsigned long long o = red[tid + off];
        if (o > red[tid]) red[tid] = o;
      }
      __syncthreads();
    }
    if (tid == 0) {
      int f = MAXNP - 1 - (int)(red[0] & 0xFFFFFFFFull);
      sfar = f;
      pool[b * NFPS + t] = f;
    }
    __syncthreads();
  }
}

// ------------------------------------------------------------------
// Final gathers: fps_indices (as float), vote_candidates, vote_features
// ------------------------------------------------------------------
__global__ __launch_bounds__(256)
void gather_kernel(const int* __restrict__ topk, const int* __restrict__ pool,
                   const float* __restrict__ votes, const float* __restrict__ feats,
                   float* __restrict__ vc_out, float* __restrict__ vf_out,
                   float* __restrict__ fidx_out) {
  int r = blockIdx.x;              // 0..4095
  int b = r >> 10, t = r & 1023;
  int p  = pool[b * NFPS + t];
  int gi = topk[b * MAXNP + p] + b * NPB;
  int tid = threadIdx.x;
  vf_out[(long)r * CDIM + tid] = feats[(long)gi * CDIM + tid];
  if (tid < 4)  vc_out[r * 4 + tid] = votes[(long)gi * 4 + tid];
  if (tid == 0) fidx_out[r] = (float)gi;
}

// ------------------------------------------------------------------
extern "C" void kernel_launch(void* const* d_in, const int* in_sizes, int n_in,
                              void* d_out, int out_size, void* d_ws, size_t ws_size,
                              hipStream_t stream) {
  const int*   voxel_coords  = (const int*)  d_in[0];
  const float* seed_features = (const float*)d_in[1];
  const int*   spatial_str   = (const int*)  d_in[3];
  const float* W_off1 = (const float*)d_in[4];
  const float* g_off  = (const float*)d_in[5];
  const float* b_off  = (const float*)d_in[6];
  const float* W_off2 = (const float*)d_in[7];
  const float* b_off2 = (const float*)d_in[8];
  const float* W_cls1 = (const float*)d_in[9];
  const float* g_cls  = (const float*)d_in[10];
  const float* b_cls  = (const float*)d_in[11];
  const float* W_cls2 = (const float*)d_in[12];
  const float* b_cls2 = (const float*)d_in[13];
  float* out = (float*)d_out;

  // workspace layout (all offsets keep >=256B alignment)
  char* ws = (char*)d_ws;
  size_t o = 0;
  bf16_t* Woff = (bf16_t*)(ws + o); o += (size_t)CDIM * CDIM * sizeof(bf16_t);
  bf16_t* Wcls = (bf16_t*)(ws + o); o += (size_t)CDIM * CDIM * sizeof(bf16_t);
  bf16_t* Hoff = (bf16_t*)(ws + o); o += (size_t)NTOT * CDIM * sizeof(bf16_t);
  bf16_t* Hcls = (bf16_t*)(ws + o); o += (size_t)NTOT * CDIM * sizeof(bf16_t);
  float*  stats = (float*)(ws + o); o += 2048 * sizeof(float);
  float*  sb    = (float*)(ws + o); o += (size_t)NTOT * sizeof(float);
  int*    topk  = (int*)  (ws + o); o += (size_t)BATCH * MAXNP * sizeof(int);
  int*    pool  = (int*)  (ws + o); o += (size_t)BATCH * NFPS * sizeof(int);

  // 1) weight bf16 conversions + stat zeroing (X converted in-register in GEMM)
  f32_to_bf16_kernel<<<(CDIM * CDIM + 255) / 256, 256, 0, stream>>>(W_off1, Woff, CDIM * CDIM);
  f32_to_bf16_kernel<<<(CDIM * CDIM + 255) / 256, 256, 0, stream>>>(W_cls1, Wcls, CDIM * CDIM);
  zero_kernel<<<4, 256, 0, stream>>>(stats, 1024);

  // 2) fused dual-head WMMA GEMM (X read once) with fused BN-statistics
  gemm_fused_kernel<<<NTOT / 128, 256, 0, stream>>>(seed_features, Woff, Wcls,
                                                    Hoff, Hcls, stats);
  finalize_stats_kernel<<<2, 256, 0, stream>>>(stats);

  // 3) seeds, then head tails -> votes / seed_cls / scores
  seeds_kernel<<<NTOT / 256, 256, 0, stream>>>(voxel_coords, spatial_str, out + OUT_SEEDS);
  head2_off_kernel<<<NTOT / 8, 256, 0, stream>>>(Hoff, stats + 1024, stats + 1280,
                                                 g_off, b_off, W_off2, b_off2,
                                                 out + OUT_SEEDS, out + OUT_VOTES);
  head2_cls_kernel<<<NTOT / 8, 256, 0, stream>>>(Hcls, stats + 1536, stats + 1792,
                                                 g_cls, b_cls, W_cls2, b_cls2,
                                                 out + OUT_SCLS, sb);

  // 4) selection pipeline
  topk_kernel<<<BATCH, 1024, 0, stream>>>(sb, topk);
  fps_kernel<<<BATCH, 1024, 0, stream>>>(topk, out + OUT_SEEDS, pool);
  gather_kernel<<<BATCH * NFPS, 256, 0, stream>>>(topk, pool, out + OUT_VOTES,
                                                  seed_features,
                                                  out + OUT_VC, out + OUT_VF,
                                                  out + OUT_FIDX);
}
